// DGCNN_32633161515820
// MI455X (gfx1250) — matrix operand order repaired and checked by reading, not compiled, not run on hardware
//
#include <hip/hip_runtime.h>
#include <hip/hip_bf16.h>

// ---------------------------------------------------------------------------
// DGCNN edge-conv block for gfx1250 (MI455X), wave32 + WMMA f16 + TDM.
// Pipeline: KNN -> gather+conv1(6->64, VALU) -> [WMMA GEMM (B tile staged to
// LDS by Tensor Data Mover) -> BN stats -> BN/ReLU/max-over-K] x3 ->
// conv5 (512x512 WMMA) -> BN/ReLU -> out.
// Activations kept in f16 point-major (m, C); WMMA accumulates in f32.
// ---------------------------------------------------------------------------

typedef _Float16     v16h  __attribute__((ext_vector_type(16)));
typedef _Float16     h8    __attribute__((ext_vector_type(8)));
typedef float        v8f   __attribute__((ext_vector_type(8)));
typedef unsigned int u32x4 __attribute__((ext_vector_type(4)));
typedef int          i32x4 __attribute__((ext_vector_type(4)));
typedef int          i32x8 __attribute__((ext_vector_type(8)));

#define NB   8
#define NPT  2048
#define KNN  32
#define M1   (NB * NPT * KNN)   // 524288 edge samples
#define NPts (NB * NPT)         // 16384 points
#define EMB  512

#if defined(__HIP_DEVICE_COMPILE__) && __has_builtin(__builtin_amdgcn_tensor_load_to_lds)
#define HAVE_TDM 1
#else
#define HAVE_TDM 0
#endif

// ------------------------------- KNN ---------------------------------------
// One 256-thread block per (b, n) row. Distance row in LDS, then 32 argmax
// sweeps (gives descending order like lax.top_k; self is first neighbor).
__global__ void knn_kernel(const float* __restrict__ x, int* __restrict__ idxf) {
    __shared__ float dist[NPT];
    __shared__ float rv[256];
    __shared__ int   ri[256];
    const int row = blockIdx.x;
    const int b = row / NPT, n = row % NPT;
    const int tid = threadIdx.x;
    const float xi0 = x[(b * 3 + 0) * NPT + n];
    const float xi1 = x[(b * 3 + 1) * NPT + n];
    const float xi2 = x[(b * 3 + 2) * NPT + n];
    for (int j = tid; j < NPT; j += 256) {
        float d0 = xi0 - x[(b * 3 + 0) * NPT + j];
        float d1 = xi1 - x[(b * 3 + 1) * NPT + j];
        float d2 = xi2 - x[(b * 3 + 2) * NPT + j];
        dist[j] = -(d0 * d0 + d1 * d1 + d2 * d2);
    }
    __syncthreads();
    for (int kk = 0; kk < KNN; ++kk) {
        float bv = -3.4e38f; int bi = 0;
        for (int j = tid; j < NPT; j += 256) {
            float v = dist[j];
            if (v > bv) { bv = v; bi = j; }
        }
        rv[tid] = bv; ri[tid] = bi;
        __syncthreads();
        for (int s = 128; s > 0; s >>= 1) {
            if (tid < s && rv[tid + s] > rv[tid]) { rv[tid] = rv[tid + s]; ri[tid] = ri[tid + s]; }
            __syncthreads();
        }
        if (tid == 0) {
            idxf[row * KNN + kk] = b * NPT + ri[0];   // flattened with batch base
            dist[ri[0]] = -3.4e38f;
        }
        __syncthreads();
    }
}

// ------------------------- weight f32 -> f16 --------------------------------
__global__ void cvt_f16_kernel(const float* __restrict__ s, _Float16* __restrict__ d, int n) {
    int i = blockIdx.x * blockDim.x + threadIdx.x;
    if (i < n) d[i] = (_Float16)s[i];
}

// ---------------- gather + conv1 (Cin=6 -> Cout=64, VALU) -------------------
// Also emits the `nbrs` output (B,N,K,3). Y1 stored f16 point-major (m,64).
__global__ void gather_conv1_kernel(const float* __restrict__ x, const int* __restrict__ idxf,
                                    const float* __restrict__ W1,
                                    float* __restrict__ nbrs, _Float16* __restrict__ Y) {
    __shared__ float w[64 * 6];
    const int tid = threadIdx.x;
    for (int i = tid; i < 64 * 6; i += 256) w[i] = W1[i];
    __syncthreads();
    const int m = blockIdx.x * 256 + tid;
    if (m >= M1) return;
    const int b = m / (NPT * KNN);
    const int n = (m / KNN) % NPT;
    const int j = idxf[m] - b * NPT;
    float f[6];
    for (int c = 0; c < 3; ++c) {
        float xj = x[(b * 3 + c) * NPT + j];
        f[c]     = xj;                        // neighbor first (concat order)
        f[3 + c] = x[(b * 3 + c) * NPT + n];  // then center
        nbrs[m * 3 + c] = xj;
    }
    for (int o = 0; o < 64; ++o) {
        float acc = 0.f;
        for (int c = 0; c < 6; ++c) acc += w[o * 6 + c] * f[c];
        Y[(size_t)m * 64 + o] = (_Float16)acc;
    }
}

// ----------------------------- WMMA GEMM ------------------------------------
// D(Cout x M) = A(Cout x Cin) * B(Cin x M); A row-major f16 (L2-resident
// weights, read from global), B = activations point-major (m, Cin) f16 staged
// ONCE per workgroup into LDS by the Tensor Data Mover; all waves in the block
// compute different 16-channel o-tiles against the shared 16-point m-tile.
// TDM pads LDS rows to Cin+8 halfs -> conflict-free ds_load_b128 fragments.
#define BPAD 8
__global__ void wmma_gemm_kernel(const _Float16* __restrict__ A, const _Float16* __restrict__ Bm,
                                 _Float16* __restrict__ Y, int Cout, int Cin, int M) {
    __shared__ __align__(16) _Float16 sB[16 * (512 + BPAD)];   // max Cin=512 -> 16.6 KB
    const int tid  = threadIdx.x;
    const int lane = tid & 31;
    const int m0   = blockIdx.x << 4;
    const int to   = blockIdx.y * (blockDim.x >> 5) + (tid >> 5);
    const int tilesO = Cout >> 4;
    const int strideB = Cin + BPAD;

#if HAVE_TDM
    if (tid < 32) {
        // --- Tensor DMA descriptor (D#): 2D tile, 16 rows x Cin/2 dwords ---
        const unsigned W = (unsigned)(Cin >> 1);            // dwords per row
        const int lg = 31 - __clz((int)W);                  // W is a power of two
        const unsigned pic = (unsigned)(lg - 1);            // pad_interval: 2<<pic dwords
        const unsigned lds_off = (unsigned)(uintptr_t)&sB[0];
        const unsigned long long ga =
            (unsigned long long)(uintptr_t)(Bm + (size_t)m0 * Cin);
        u32x4 g0;
        g0.x = 1u;                                          // count = 1 valid descriptor
        g0.y = lds_off;                                     // LDS destination (bytes)
        g0.z = (unsigned)ga;                                // global_addr[31:0]
        g0.w = (unsigned)((ga >> 32) & 0x1FFFFFFull) | (2u << 30);  // addr[56:32] | type=2
        i32x8 g1;
        g1[0] = (int)((2u << 16) | (1u << 20) | (pic << 22) | (3u << 25)); // 4B elems, pad_en,
                                                            // interval=row, amount=4 dwords
        g1[1] = (int)(W << 16);                             // tensor_dim0[15:0]
        g1[2] = (int)(0xFFFFu << 16);                       // dim0 hi=0 | tensor_dim1 lo
        g1[3] = (int)(0x0000FFFFu | (W << 16));             // tensor_dim1 hi | tile_dim0=W
        g1[4] = 16;                                         // tile_dim1=16, tile_dim2=0
        g1[5] = (int)W;                                     // tensor_dim0_stride (contiguous)
        g1[6] = 0;
        g1[7] = 0;
        i32x4 z4 = {};
#if defined(__clang_major__) && __clang_major__ >= 23
        i32x8 z8 = {};
        __builtin_amdgcn_tensor_load_to_lds(g0, g1, z4, z4, z8, 0);
#else
        __builtin_amdgcn_tensor_load_to_lds(g0, g1, z4, z4, 0);
#endif
        __builtin_amdgcn_s_wait_tensorcnt(0);
    }
#else
    // Fallback: cooperative staging with the same padded layout.
    for (int i = tid; i < (Cin >> 3) * 16; i += blockDim.x) {
        const int r = i / (Cin >> 3);
        const int c = (i % (Cin >> 3)) << 3;
        *(h8*)&sB[r * strideB + c] = *(const h8*)(Bm + (size_t)(m0 + r) * Cin + c);
    }
#endif
    __syncthreads();

    if (to < tilesO) {                       // wave-uniform: EXEC all-1 for WMMA
        const int o0 = to << 4;
        const int lr = lane & 15;
        const bool hi = lane >= 16;
        // ISA 16-bit A 16x32 layout: lo lanes K{0..7,16..23}, hi lanes K{8..15,24..31}
        const _Float16* arow = A + (size_t)(o0 + lr) * Cin + (hi ? 8 : 0);
        // ISA 16-bit B 32x16 layout: lo lanes K{0..7,16..23}, hi lanes K{8..15,24..31}
        const int bbase = lr * strideB + (hi ? 8 : 0);
        v8f c = {};
        for (int k = 0; k < Cin; k += 32) {
            h8 a0 = *(const h8*)(arow + k);
            h8 a1 = *(const h8*)(arow + k + 16);
            h8 b0 = *(const h8*)&sB[bbase + k];
            h8 b1 = *(const h8*)&sB[bbase + k + 16];
            v16h av = __builtin_shufflevector(a0, a1, 0,1,2,3,4,5,6,7,8,9,10,11,12,13,14,15);
            v16h bv = __builtin_shufflevector(b0, b1, 0,1,2,3,4,5,6,7,8,9,10,11,12,13,14,15);
            c = __builtin_amdgcn_wmma_f32_16x16x32_f16(false, av, false, bv, (short)0, c, false, false);
        }
        // D layout: lane -> point m0+lr, 8 consecutive channels (lo: o0.., hi: o0+8..)
        h8 pk;
        for (int r = 0; r < 8; ++r) pk[r] = (_Float16)c[r];
        *(h8*)(Y + (size_t)(m0 + lr) * Cout + o0 + (hi ? 8 : 0)) = pk;
    }
}

// ------------------------- BN batch statistics ------------------------------
#define STATS_CHUNK 2048
__global__ void stats_kernel(const _Float16* __restrict__ Y, float* __restrict__ sum,
                             float* __restrict__ sumsq, int Cout, int M) {
    const int o = blockIdx.y * 64 + threadIdx.x;
    const int m0 = blockIdx.x * STATS_CHUNK;
    const int m1 = min(M, m0 + STATS_CHUNK);
    float s = 0.f, q = 0.f;
    for (int m = m0; m < m1; ++m) {
        float v = (float)Y[(size_t)m * Cout + o];
        s += v; q += v * v;
    }
    atomicAdd(&sum[o], s);
    atomicAdd(&sumsq[o], q);
}

__global__ void bnparams_kernel(const float* __restrict__ sum, const float* __restrict__ sumsq,
                                const float* __restrict__ g, const float* __restrict__ bb,
                                float* __restrict__ scale, float* __restrict__ shift, int M) {
    const int o = threadIdx.x;
    const float inv = 1.0f / (float)M;
    const float mu  = sum[o] * inv;
    const float var = fmaxf(sumsq[o] * inv - mu * mu, 0.f);
    const float rstd = rsqrtf(var + 1e-5f);
    const float sc = g[o] * rstd;
    scale[o] = sc;
    shift[o] = bb[o] - mu * sc;
}

// --------------- BN apply + ReLU + max over K (layers 1..4) -----------------
// Writes f16 H (next GEMM input) and the max-pooled slice into cat(p, 512).
__global__ void apply_max_kernel(const _Float16* __restrict__ Y, const float* __restrict__ scale,
                                 const float* __restrict__ shift, _Float16* __restrict__ H,
                                 _Float16* __restrict__ cat, int catoff, int Cout) {
    const int p = blockIdx.x;
    const int o = blockIdx.y * 64 + threadIdx.x;
    const float sc = scale[o], sh = shift[o];
    float mx = 0.f;
    for (int k = 0; k < KNN; ++k) {
        const size_t m = (size_t)p * KNN + k;
        float v = (float)Y[m * Cout + o] * sc + sh;
        v = fmaxf(v, 0.f);
        if (H) H[m * Cout + o] = (_Float16)v;
        mx = fmaxf(mx, v);
    }
    cat[(size_t)p * EMB + catoff + o] = (_Float16)mx;
}

// --------------------- final BN + ReLU, (B,512,N) layout --------------------
__global__ void apply_out_kernel(const _Float16* __restrict__ Y5, const float* __restrict__ scale,
                                 const float* __restrict__ shift, float* __restrict__ out) {
    const int p = blockIdx.x;
    const int o = blockIdx.y * 64 + threadIdx.x;
    const int b = p / NPT, n = p % NPT;
    float v = (float)Y5[(size_t)p * EMB + o] * scale[o] + shift[o];
    out[((size_t)b * EMB + o) * NPT + n] = fmaxf(v, 0.f);
}

// ---------------------------------------------------------------------------
extern "C" void kernel_launch(void* const* d_in, const int* in_sizes, int n_in,
                              void* d_out, int out_size, void* d_ws, size_t ws_size,
                              hipStream_t stream) {
    const float* x  = (const float*)d_in[0];
    const float* W1 = (const float*)d_in[1];
    const float* g1 = (const float*)d_in[2];  const float* b1 = (const float*)d_in[3];
    const float* W2 = (const float*)d_in[4];
    const float* g2 = (const float*)d_in[5];  const float* b2 = (const float*)d_in[6];
    const float* W3 = (const float*)d_in[7];
    const float* g3 = (const float*)d_in[8];  const float* b3 = (const float*)d_in[9];
    const float* W4 = (const float*)d_in[10];
    const float* g4 = (const float*)d_in[11]; const float* b4 = (const float*)d_in[12];
    const float* W5 = (const float*)d_in[13];
    const float* g5 = (const float*)d_in[14]; const float* b5 = (const float*)d_in[15];

    // d_out = [out (8*512*2048 f32) | idx_flat (524288 i32) | nbrs (8*2048*32*3 f32)]
    float* out  = (float*)d_out;
    int*   idxf = (int*)(out + (size_t)NB * EMB * NPT);
    float* nbrs = (float*)(idxf + (size_t)M1);

    // Workspace layout (f16 activations; Y reused across layers).
    char* w = (char*)d_ws;
    _Float16* Y   = (_Float16*)w;                       w += (size_t)M1 * 256 * 2; // 268 MB max (layer4)
    _Float16* HA  = (_Float16*)w;                       w += (size_t)M1 * 128 * 2; // 134 MB
    _Float16* HB  = (_Float16*)w;                       w += (size_t)M1 * 128 * 2; // 134 MB
    _Float16* CAT = (_Float16*)w;                       w += (size_t)NPts * EMB * 2;
    _Float16* W2h = (_Float16*)w;                       w += 64 * 64 * 2;
    _Float16* W3h = (_Float16*)w;                       w += 128 * 64 * 2;
    _Float16* W4h = (_Float16*)w;                       w += 256 * 128 * 2;
    _Float16* W5h = (_Float16*)w;                       w += 512 * 512 * 2;
    float* sum   = (float*)w;                           w += 512 * 4;
    float* sumsq = (float*)w;                           w += 512 * 4;
    float* scale = (float*)w;                           w += 512 * 4;
    float* shift = (float*)w;                           w += 512 * 4;

    // 1) KNN (also writes idx_flat output)
    knn_kernel<<<NPts, 256, 0, stream>>>(x, idxf);

    // 2) weights to f16
    cvt_f16_kernel<<<(64 * 64 + 255) / 256, 256, 0, stream>>>(W2, W2h, 64 * 64);
    cvt_f16_kernel<<<(128 * 64 + 255) / 256, 256, 0, stream>>>(W3, W3h, 128 * 64);
    cvt_f16_kernel<<<(256 * 128 + 255) / 256, 256, 0, stream>>>(W4, W4h, 256 * 128);
    cvt_f16_kernel<<<(512 * 512 + 255) / 256, 256, 0, stream>>>(W5, W5h, 512 * 512);

    // 3) gather + conv1 (also writes nbrs output)
    gather_conv1_kernel<<<M1 / 256, 256, 0, stream>>>(x, idxf, W1, nbrs, Y);

    auto bn_stage = [&](int Cout, int M, const float* g, const float* b) {
        hipMemsetAsync(sum, 0, 2 * 512 * sizeof(float), stream);  // sum+sumsq adjacent
        stats_kernel<<<dim3(M / STATS_CHUNK, Cout / 64), 64, 0, stream>>>(Y, sum, sumsq, Cout, M);
        bnparams_kernel<<<1, Cout, 0, stream>>>(sum, sumsq, g, b, scale, shift, M);
    };
    auto gemm = [&](const _Float16* A, const _Float16* Bm, int Cout, int Cin, int M) {
        const int tilesO = Cout / 16;
        const int wpb = tilesO < 8 ? tilesO : 8;            // waves per block
        wmma_gemm_kernel<<<dim3(M / 16, tilesO / wpb), wpb * 32, 0, stream>>>(A, Bm, Y, Cout, Cin, M);
    };

    // layer 1: BN/ReLU/max -> HA (m,64), cat[0:64]
    bn_stage(64, M1, g1, b1);
    apply_max_kernel<<<dim3(NPts, 1), 64, 0, stream>>>(Y, scale, shift, HA, CAT, 0, 64);

    // layer 2: 64->64
    gemm(W2h, HA, 64, 64, M1);
    bn_stage(64, M1, g2, b2);
    apply_max_kernel<<<dim3(NPts, 1), 64, 0, stream>>>(Y, scale, shift, HB, CAT, 64, 64);

    // layer 3: 64->128
    gemm(W3h, HB, 128, 64, M1);
    bn_stage(128, M1, g3, b3);
    apply_max_kernel<<<dim3(NPts, 2), 64, 0, stream>>>(Y, scale, shift, HA, CAT, 128, 128);

    // layer 4: 128->256 (no H needed, only pooled x4)
    gemm(W4h, HA, 256, 128, M1);
    bn_stage(256, M1, g4, b4);
    apply_max_kernel<<<dim3(NPts, 4), 64, 0, stream>>>(Y, scale, shift, (_Float16*)nullptr, CAT, 256, 256);

    // layer 5: 512->512 over pooled points
    gemm(W5h, CAT, 512, 512, NPts);
    bn_stage(512, NPts, g5, b5);
    apply_out_kernel<<<dim3(NPts, 8), 64, 0, stream>>>(Y, scale, shift, out);
}
